// SGCNs_59940563583095
// MI455X (gfx1250) — matrix operand order repaired and checked by reading, not compiled, not run on hardware
//
#include <hip/hip_runtime.h>

#define B 32
#define CIN 512
#define K 1024
#define CMID 128
#define COUT 64
#define NCLS 5
#define NUM_TOP 205
#define NUM_KEPT 409
#define NUM_DROP 410
#define ADJK 204
#define EPSF 1e-8f
#define INV_DROP_SCALE (1.0f / (1.0f - 410.0f / 1024.0f))

typedef __attribute__((ext_vector_type(16))) _Float16 v16h;
typedef __attribute__((ext_vector_type(8)))  _Float16 v8h;
typedef __attribute__((ext_vector_type(8)))  float    v8f;

union F16x16 { _Float16 h[16]; v8h v8[2]; v16h v; };

struct Acc22 { v8f a[2][2]; };

#define LDSA_STRIDE 40   // 80B row stride: multiple of 16B, conflict-friendly
#define LDSB_STRIDE 72   // 144B row stride for [d][n] tiles

// ---------------------------------------------------------------------------
// GEMM cores: 128 threads (4 waves) compute a 64x64 f32 tile, chunked depth 32.
// Wave w -> (wm=w>>1, wn=w&1) quadrant, each wave does 2x2 WMMAs.
// ---------------------------------------------------------------------------

// NT: A(m,d)=A64[m*lda+d], B(n,d)=B64[n*ldb+d]  (both unit-stride along depth)
__device__ __forceinline__ void gemm_core_nt(
    const _Float16* __restrict__ A64, int lda,
    const _Float16* __restrict__ B64, int ldb,
    int depth, _Float16* As, _Float16* Bs, Acc22& acc)
{
  const int tid = threadIdx.x;
  const int lane = tid & 31, wave = tid >> 5;
  const int wm = wave >> 1, wn = wave & 1;
  const int mrow = lane & 15, hi = lane >> 4;
  for (int d0 = 0; d0 < depth; d0 += 32) {
    {
      int r = tid >> 1, hf = (tid & 1) * 16;
      const v8h* ga = (const v8h*)(A64 + (size_t)r * lda + d0 + hf);
      v8h* la = (v8h*)(As + r * LDSA_STRIDE + hf);
      la[0] = ga[0]; la[1] = ga[1];
      const v8h* gb = (const v8h*)(B64 + (size_t)r * ldb + d0 + hf);
      v8h* lb = (v8h*)(Bs + r * LDSA_STRIDE + hf);
      lb[0] = gb[0]; lb[1] = gb[1];
      if (d0 + 32 < depth) {
        __builtin_prefetch(A64 + (size_t)r * lda + d0 + 32 + hf, 0, 0);
        __builtin_prefetch(B64 + (size_t)r * ldb + d0 + 32 + hf, 0, 0);
      }
    }
    __syncthreads();
    F16x16 afr[2], bfr[2];
    for (int i = 0; i < 2; ++i) {
      int m = wm * 32 + i * 16 + mrow;
      afr[i].v8[0] = *(const v8h*)(As + m * LDSA_STRIDE + hi * 8);
      afr[i].v8[1] = *(const v8h*)(As + m * LDSA_STRIDE + 16 + hi * 8);
      int n = wn * 32 + i * 16 + mrow;
      bfr[i].v8[0] = *(const v8h*)(Bs + n * LDSA_STRIDE + hi * 8);
      bfr[i].v8[1] = *(const v8h*)(Bs + n * LDSA_STRIDE + 16 + hi * 8);
    }
    for (int i = 0; i < 2; ++i)
      for (int j = 0; j < 2; ++j)
        acc.a[i][j] = __builtin_amdgcn_wmma_f32_16x16x32_f16(
            false, afr[i].v, false, bfr[j].v, (short)0, acc.a[i][j], false, false);
    __syncthreads();
  }
}

// NN: A(m,d)=A64[m*lda+d] (unit depth), B(d,n)=Bdn[d*ldb+n] (unit n, 64 cols)
__device__ __forceinline__ void gemm_core_nn(
    const _Float16* __restrict__ A64, int lda,
    const _Float16* __restrict__ Bdn, int ldb,
    int depth, _Float16* As, _Float16* Bs, Acc22& acc)
{
  const int tid = threadIdx.x;
  const int lane = tid & 31, wave = tid >> 5;
  const int wm = wave >> 1, wn = wave & 1;
  const int mrow = lane & 15, hi = lane >> 4;
  for (int d0 = 0; d0 < depth; d0 += 32) {
    {
      int r = tid >> 1, hf = (tid & 1) * 16;
      const v8h* ga = (const v8h*)(A64 + (size_t)r * lda + d0 + hf);
      v8h* la = (v8h*)(As + r * LDSA_STRIDE + hf);
      la[0] = ga[0]; la[1] = ga[1];
      int d = tid >> 2, n8 = (tid & 3) * 16;
      const v8h* gb = (const v8h*)(Bdn + (size_t)(d0 + d) * ldb + n8);
      v8h* lb = (v8h*)(Bs + d * LDSB_STRIDE + n8);
      lb[0] = gb[0]; lb[1] = gb[1];
      if (d0 + 32 < depth) {
        __builtin_prefetch(A64 + (size_t)r * lda + d0 + 32 + hf, 0, 0);
        __builtin_prefetch(Bdn + (size_t)(d0 + 32 + d) * ldb + n8, 0, 0);
      }
    }
    __syncthreads();
    F16x16 afr[2], bfr[2];
    for (int i = 0; i < 2; ++i) {
      int m = wm * 32 + i * 16 + mrow;
      afr[i].v8[0] = *(const v8h*)(As + m * LDSA_STRIDE + hi * 8);
      afr[i].v8[1] = *(const v8h*)(As + m * LDSA_STRIDE + 16 + hi * 8);
    }
    for (int j = 0; j < 2; ++j) {
      int n = wn * 32 + j * 16 + mrow;
      for (int v = 0; v < 8; ++v) {
        int kk = hi * 8 + 2 * v + (v >= 4 ? 8 : 0);
        bfr[j].h[2 * v]     = Bs[kk * LDSB_STRIDE + n];
        bfr[j].h[2 * v + 1] = Bs[(kk + 1) * LDSB_STRIDE + n];
      }
    }
    for (int i = 0; i < 2; ++i)
      for (int j = 0; j < 2; ++j)
        acc.a[i][j] = __builtin_amdgcn_wmma_f32_16x16x32_f16(
            false, afr[i].v, false, bfr[j].v, (short)0, acc.a[i][j], false, false);
    __syncthreads();
  }
}

__device__ __forceinline__ void acc_zero(Acc22& acc) {
  v8f z = {0.f, 0.f, 0.f, 0.f, 0.f, 0.f, 0.f, 0.f};
  acc.a[0][0] = z; acc.a[0][1] = z; acc.a[1][0] = z; acc.a[1][1] = z;
}

__device__ __forceinline__ float block_reduce(float s, float* sh) {
  int t = threadIdx.x;
  sh[t] = s; __syncthreads();
  for (int off = blockDim.x >> 1; off > 0; off >>= 1) {
    if (t < off) sh[t] += sh[t + off];
    __syncthreads();
  }
  float r = sh[0]; __syncthreads();
  return r;
}

__device__ __forceinline__ unsigned fkey(float f) {
  unsigned u = __float_as_uint(f);
  return (u & 0x80000000u) ? ~u : (u | 0x80000000u);  // monotonic float order
}

// ---------------------------------------------------------------------------
// 1) SGA masks: exact rank-by-counting (matches jax top_k index tie-break)
// ---------------------------------------------------------------------------
__global__ void k_masks(const float* __restrict__ imp, const float* __restrict__ rnd,
                        float* topm, float* keptm, float* dropm, float* bmask, float* kms) {
  __shared__ float v[K];
  int b = blockIdx.x, i = threadIdx.x;
  float im = imp[b * K + i];
  float rn = rnd[b * K + i];
  v[i] = im; __syncthreads();
  int cnt = 0;
  for (int j = 0; j < K; ++j) { float vj = v[j]; cnt += (vj > im) || (vj == im && j < i); }
  float top = (cnt < NUM_TOP) ? 1.f : 0.f;
  __syncthreads();
  float rm1 = (top > 0.f) ? -10.f : rn;
  v[i] = rm1; __syncthreads();
  cnt = 0;
  for (int j = 0; j < K; ++j) { float vj = v[j]; cnt += (vj > rm1) || (vj == rm1 && j < i); }
  float kept = (cnt < NUM_KEPT) ? 1.f : 0.f;
  __syncthreads();
  float rm2 = (top > 0.f) ? 10.f : rn;
  v[i] = rm2; __syncthreads();
  cnt = 0;
  for (int j = 0; j < K; ++j) { float vj = v[j]; cnt += (vj < rm2) || (vj == rm2 && j < i); }
  float drop = (cnt < NUM_DROP) ? 1.f : 0.f;
  topm[b * K + i] = top;
  keptm[b * K + i] = kept;
  dropm[b * K + i] = drop;
  bmask[b * K + i] = fminf(top + kept, 1.f);
  kms[b * K + i] = fminf(top + kept, 1.f) * INV_DROP_SCALE;
}

// ---------------------------------------------------------------------------
// 2) casts / norms / transposes
// ---------------------------------------------------------------------------
__global__ void k_cast_wT(const float* W1, const float* W2, _Float16* W1t, _Float16* W2t) {
  int idx = blockIdx.x * 256 + threadIdx.x;
  if (idx < CIN * CMID) { int c = idx / CMID, o = idx % CMID; W1t[o * CIN + c] = (_Float16)W1[idx]; }
  if (idx < CMID * COUT) { int c = idx / COUT, o = idx % COUT; W2t[o * CMID + c] = (_Float16)W2[idx]; }
}

__global__ void k_colnorm(const float* __restrict__ X, float* invn, int Cd) {
  int idx = blockIdx.x * 256 + threadIdx.x;           // b*K + k
  int b = idx >> 10, k = idx & (K - 1);
  const float* p = X + (size_t)b * Cd * K + k;
  float s = 0.f;
  for (int c = 0; c < Cd; ++c) { float f = p[(size_t)c * K]; s += f * f; }
  invn[idx] = 1.f / (sqrtf(s) + EPSF);
}

// X [b][Cd][K] f32  ->  outA[b][K][Cd] = f16(X^T * sA[b,k]),  outB same with sB
// (sA/sB == nullptr means scale 1)
__global__ void k_cast_T(const float* __restrict__ X, const float* __restrict__ sA,
                         const float* __restrict__ sB, _Float16* outA, _Float16* outB, int Cd) {
  __shared__ float tile[32][33];
  int k0 = blockIdx.x * 32, c0 = blockIdx.y * 32, b = blockIdx.z;
  int tx = threadIdx.x, ty = threadIdx.y;
  const float* Xb = X + (size_t)b * Cd * K;
  for (int r = 0; r < 4; ++r) {
    int c = c0 + ty + r * 8;
    tile[ty + r * 8][tx] = Xb[(size_t)c * K + k0 + tx];
  }
  __syncthreads();
  for (int r = 0; r < 4; ++r) {
    int k = k0 + ty + r * 8;
    int c = c0 + tx;
    float f = tile[tx][ty + r * 8];
    float a = sA ? sA[b * K + k] : 1.f;
    outA[((size_t)b * K + k) * Cd + c] = (_Float16)(f * a);
    if (outB) {
      float s2 = sB ? sB[b * K + k] : 1.f;
      outB[((size_t)b * K + k) * Cd + c] = (_Float16)(f * s2);
    }
  }
}

// ---------------------------------------------------------------------------
// 3) adjP rows (Gaussian prior, row-normalized) -> f16
// ---------------------------------------------------------------------------
__global__ void k_adjP(const float* __restrict__ imp, _Float16* adjPh) {
  __shared__ float sh[256];
  int row = blockIdx.x;                // b*K + i
  int b = row >> 10, i = row & (K - 1);
  float di = imp[b * K + i];
  float s = 0.f;
  for (int j = threadIdx.x; j < K; j += 256) {
    float d = di - imp[b * K + j];
    float e = __expf(-d * d * 0.125f);          // 2*sigma^2 = 8
    if (j == i) e += 1.f;
    s += e;
  }
  float tot = block_reduce(s, sh);
  float inv = 1.f / (tot + EPSF);
  _Float16* o = adjPh + (size_t)row * K;
  for (int j = threadIdx.x; j < K; j += 256) {
    float d = di - imp[b * K + j];
    float e = __expf(-d * d * 0.125f);
    if (j == i) e += 1.f;
    o[j] = (_Float16)(e * inv);
  }
}

// ---------------------------------------------------------------------------
// 4) WMMA GEMMs (64x64 block tiles)
// ---------------------------------------------------------------------------
__global__ void k_gemm_sim(const _Float16* __restrict__ nT, float* __restrict__ sim, int Cd) {
  __shared__ _Float16 As[64 * LDSA_STRIDE];
  __shared__ _Float16 Bs[64 * LDSA_STRIDE];
  int tn = blockIdx.x, tm = blockIdx.y, b = blockIdx.z;
  const _Float16* base = nT + (size_t)b * K * Cd;
  Acc22 acc; acc_zero(acc);
  gemm_core_nt(base + (size_t)tm * 64 * Cd, Cd, base + (size_t)tn * 64 * Cd, Cd, Cd, As, Bs, acc);
  int tid = threadIdx.x, lane = tid & 31, wave = tid >> 5;
  int wm = wave >> 1, wn = wave & 1, col = lane & 15, hi = lane >> 4;
  float* out = sim + (size_t)b * K * K;
  for (int i = 0; i < 2; ++i)
    for (int j = 0; j < 2; ++j)
      for (int v = 0; v < 8; ++v) {
        int r = tm * 64 + wm * 32 + i * 16 + v + 8 * hi;
        int c = tn * 64 + wn * 32 + j * 16 + col;
        out[(size_t)r * K + c] = acc.a[i][j][v];
      }
}

// H[m][k] = sum_d Wt[m][d] * XT[k][d]   (out f16 [M][K])
__global__ void k_gemm_h(const _Float16* __restrict__ Wt, const _Float16* __restrict__ XT,
                         _Float16* __restrict__ Hh, int M, int Cd) {
  __shared__ _Float16 As[64 * LDSA_STRIDE];
  __shared__ _Float16 Bs[64 * LDSA_STRIDE];
  int tn = blockIdx.x, tm = blockIdx.y, b = blockIdx.z;
  Acc22 acc; acc_zero(acc);
  gemm_core_nt(Wt + (size_t)tm * 64 * Cd, Cd,
               XT + (size_t)b * K * Cd + (size_t)tn * 64 * Cd, Cd, Cd, As, Bs, acc);
  int tid = threadIdx.x, lane = tid & 31, wave = tid >> 5;
  int wm = wave >> 1, wn = wave & 1, col = lane & 15, hi = lane >> 4;
  _Float16* out = Hh + (size_t)b * M * K;
  for (int i = 0; i < 2; ++i)
    for (int j = 0; j < 2; ++j)
      for (int v = 0; v < 8; ++v) {
        int r = tm * 64 + wm * 32 + i * 16 + v + 8 * hi;
        int c = tn * 64 + wn * 32 + j * 16 + col;
        out[(size_t)r * K + c] = (_Float16)acc.a[i][j][v];
      }
}

// E[m][j] = relu( sum_k H[m][k] * Asum[k][j] )   (out f32)
__global__ void k_gemm_emb1(const _Float16* __restrict__ Hh, const _Float16* __restrict__ Ah,
                            float* __restrict__ E, int M) {
  __shared__ _Float16 As[64 * LDSA_STRIDE];
  __shared__ _Float16 Bs[32 * LDSB_STRIDE];
  int tn = blockIdx.x, tm = blockIdx.y, b = blockIdx.z;
  Acc22 acc; acc_zero(acc);
  gemm_core_nn(Hh + (size_t)b * M * K + (size_t)tm * 64 * K, K,
               Ah + (size_t)b * K * K + tn * 64, K, K, As, Bs, acc);
  int tid = threadIdx.x, lane = tid & 31, wave = tid >> 5;
  int wm = wave >> 1, wn = wave & 1, col = lane & 15, hi = lane >> 4;
  float* out = E + (size_t)b * M * K;
  for (int i = 0; i < 2; ++i)
    for (int j = 0; j < 2; ++j)
      for (int v = 0; v < 8; ++v) {
        int r = tm * 64 + wm * 32 + i * 16 + v + 8 * hi;
        int c = tn * 64 + wn * 32 + j * 16 + col;
        out[(size_t)r * K + c] = fmaxf(acc.a[i][j][v], 0.f);
      }
}

// Ew[m][j] = relu(sum_k H A) * bmask[j] * imp[j]
__global__ void k_gemm_emb2(const _Float16* __restrict__ Hh, const _Float16* __restrict__ Ah,
                            const float* __restrict__ bmask, const float* __restrict__ imp,
                            float* __restrict__ Ew, int M) {
  __shared__ _Float16 As[64 * LDSA_STRIDE];
  __shared__ _Float16 Bs[32 * LDSB_STRIDE];
  int tn = blockIdx.x, tm = blockIdx.y, b = blockIdx.z;
  Acc22 acc; acc_zero(acc);
  gemm_core_nn(Hh + (size_t)b * M * K + (size_t)tm * 64 * K, K,
               Ah + (size_t)b * K * K + tn * 64, K, K, As, Bs, acc);
  int tid = threadIdx.x, lane = tid & 31, wave = tid >> 5;
  int wm = wave >> 1, wn = wave & 1, col = lane & 15, hi = lane >> 4;
  float* out = Ew + (size_t)b * M * K;
  for (int i = 0; i < 2; ++i)
    for (int j = 0; j < 2; ++j) {
      int c = tn * 64 + wn * 32 + j * 16 + col;
      float f = bmask[b * K + c] * imp[b * K + c];
      for (int v = 0; v < 8; ++v) {
        int r = tm * 64 + wm * 32 + i * 16 + v + 8 * hi;
        out[(size_t)r * K + c] = fmaxf(acc.a[i][j][v], 0.f) * f;
      }
    }
}

// ---------------------------------------------------------------------------
// 5) per-row exact k-th largest threshold + normalize + add adjP -> Asum f16
// ---------------------------------------------------------------------------
__global__ void k_thr_asum(const float* __restrict__ sim, const _Float16* __restrict__ adjPh,
                           _Float16* __restrict__ Asumh) {
  int wid = threadIdx.x >> 5, lane = threadIdx.x & 31;
  int row = blockIdx.x * 8 + wid;               // b*K + i
  int i = row & (K - 1);
  const float* srow = sim + (size_t)row * K;
  float v[32];
  for (int t = 0; t < 32; ++t) v[t] = srow[t * 32 + lane];

  // binary search on ordered-uint keys: largest T with count(key>=T) >= ADJK
  unsigned lo = 0u, hi = 0xFFFFFFFFu;
  while (lo < hi) {
    unsigned mid = lo + ((hi - lo) >> 1) + 1u;
    int c = 0;
    for (int t = 0; t < 32; ++t) c += (fkey(v[t]) >= mid) ? 1 : 0;
    for (int off = 16; off > 0; off >>= 1) c += __shfl_xor(c, off, 32);
    if (c >= ADJK) lo = mid; else hi = mid - 1u;
  }
  unsigned T = lo;

  float s = 0.f;
  for (int t = 0; t < 32; ++t) {
    float x = (fkey(v[t]) >= T) ? v[t] : 0.f;
    int j = t * 32 + lane;
    if (j == i) x += 1.f;                       // + eye AFTER threshold
    v[t] = x;
    s += x;
  }
  for (int off = 16; off > 0; off >>= 1) s += __shfl_xor(s, off, 32);
  float inv = 1.f / (s + EPSF);
  const _Float16* prow = adjPh + (size_t)row * K;
  _Float16* orow = Asumh + (size_t)row * K;
  for (int t = 0; t < 32; ++t) {
    int j = t * 32 + lane;
    orow[j] = (_Float16)(v[t] * inv + (float)prow[j]);
  }
}

// ---------------------------------------------------------------------------
// 6) reductions and head
// ---------------------------------------------------------------------------
__global__ void k_bags(const float* __restrict__ E, const float* __restrict__ imp,
                       const float* __restrict__ topm, const float* __restrict__ keptm,
                       const float* __restrict__ dropm, float* bagk, float* bagd) {
  __shared__ float sh[256];
  int b = blockIdx.x >> 7, c = blockIdx.x & (CMID - 1);
  const float* e = E + ((size_t)b * CMID + c) * K;
  float sk = 0.f, sd = 0.f;
  for (int k = threadIdx.x; k < K; k += 256) {
    float f = e[k] * imp[b * K + k] * INV_DROP_SCALE;
    sk += f * fminf(topm[b * K + k] + keptm[b * K + k], 1.f);
    sd += f * fminf(topm[b * K + k] + dropm[b * K + k], 1.f);
  }
  float tk = block_reduce(sk, sh);
  float td = block_reduce(sd, sh);
  if (threadIdx.x == 0) { bagk[b * CMID + c] = tk; bagd[b * CMID + c] = td; }
}

__global__ void k_bagred(const float* __restrict__ Ew, float* bag) {
  __shared__ float sh[256];
  int b = blockIdx.x / COUT, o = blockIdx.x % COUT;
  const float* e = Ew + ((size_t)b * COUT + o) * K;
  float s = 0.f;
  for (int k = threadIdx.x; k < K; k += 256) s += e[k];
  float t = block_reduce(s, sh);
  if (threadIdx.x == 0) bag[b * COUT + o] = t;
}

__global__ void k_pooled(const float* __restrict__ X, const float* __restrict__ imp,
                         float* pooled) {
  __shared__ float sh[256];
  int b = blockIdx.x / CIN, c = blockIdx.x % CIN;
  const float* x = X + ((size_t)b * CIN + c) * K;
  float s = 0.f;
  for (int k = threadIdx.x; k < K; k += 256) s += x[k] * imp[b * K + k];
  float t = block_reduce(s, sh);
  if (threadIdx.x == 0) pooled[b * CIN + c] = t;
}

__global__ void k_head(const float* __restrict__ pooled, const float* __restrict__ bag,
                       const float* __restrict__ idw, const float* __restrict__ idb,
                       const float* __restrict__ fcw, float* out) {
  __shared__ float t[COUT];
  int b = blockIdx.x, o = threadIdx.x;
  float s = idb[o];
  const float* p = pooled + b * CIN;
  for (int c = 0; c < CIN; ++c) s += p[c] * idw[c * COUT + o];
  t[o] = fmaxf(s, 0.f) + bag[b * COUT + o];
  __syncthreads();
  if (o < NCLS) {
    float a = 0.f;
    for (int q = 0; q < COUT; ++q) a += t[q] * fcw[q * NCLS + o];
    out[b * NCLS + o] = a;
  }
}

// ---------------------------------------------------------------------------
extern "C" void kernel_launch(void* const* d_in, const int* in_sizes, int n_in,
                              void* d_out, int out_size, void* d_ws, size_t ws_size,
                              hipStream_t stream) {
  const float* x   = (const float*)d_in[0];
  const float* imp = (const float*)d_in[1];
  const float* rnd = (const float*)d_in[2];
  const float* W1  = (const float*)d_in[3];
  const float* W2  = (const float*)d_in[4];
  const float* fcw = (const float*)d_in[5];
  const float* idw = (const float*)d_in[6];
  const float* idb = (const float*)d_in[7];
  (void)in_sizes; (void)n_in; (void)out_size; (void)ws_size;

  float* out  = (float*)d_out;
  float* emb1 = out + B * NCLS;                         // [B,CMID,K] f32
  float* bagk = emb1 + (size_t)B * CMID * K;            // [B,CMID]
  float* bagd = bagk + (size_t)B * CMID;                // [B,CMID]

  char* p = (char*)d_ws;
  auto alloc = [&](size_t bytes) -> char* {
    char* r = p; p += (bytes + 255) & ~(size_t)255; return r;
  };
  _Float16* xT    = (_Float16*)alloc((size_t)B * K * CIN * 2);   // x^T      [b][k][c]
  _Float16* nT1   = (_Float16*)alloc((size_t)B * K * CIN * 2);   // norm x^T
  _Float16* adjPh = (_Float16*)alloc((size_t)B * K * K * 2);
  float*    sim   = (float*)   alloc((size_t)B * K * K * 4);
  _Float16* Asumh = (_Float16*)alloc((size_t)B * K * K * 2);
  _Float16* h1h   = (_Float16*)alloc((size_t)B * CMID * K * 2);  // [b][o][k]
  _Float16* n2T   = (_Float16*)alloc((size_t)B * K * CMID * 2);  // norm emb1^T
  _Float16* e1kT  = (_Float16*)alloc((size_t)B * K * CMID * 2);  // kept emb1^T
  _Float16* h2h   = (_Float16*)alloc((size_t)B * COUT * K * 2);
  float*    em2w  = (float*)   alloc((size_t)B * COUT * K * 4);
  _Float16* W1t   = (_Float16*)alloc((size_t)CMID * CIN * 2);
  _Float16* W2t   = (_Float16*)alloc((size_t)COUT * CMID * 2);
  float*    invn1 = (float*)   alloc((size_t)B * K * 4);
  float*    invn2 = (float*)   alloc((size_t)B * K * 4);
  float*    topm  = (float*)   alloc((size_t)B * K * 4);
  float*    keptm = (float*)   alloc((size_t)B * K * 4);
  float*    dropm = (float*)   alloc((size_t)B * K * 4);
  float*    bmask = (float*)   alloc((size_t)B * K * 4);
  float*    kms   = (float*)   alloc((size_t)B * K * 4);
  float*    bag   = (float*)   alloc((size_t)B * COUT * 4);
  float*    pooled= (float*)   alloc((size_t)B * CIN * 4);

  // masks + casts + priors
  k_masks<<<B, K, 0, stream>>>(imp, rnd, topm, keptm, dropm, bmask, kms);
  k_cast_wT<<<(CIN * CMID + 255) / 256, 256, 0, stream>>>(W1, W2, W1t, W2t);
  k_colnorm<<<(B * K) / 256, 256, 0, stream>>>(x, invn1, CIN);
  k_cast_T<<<dim3(K / 32, CIN / 32, B), dim3(32, 8), 0, stream>>>(x, invn1, nullptr, nT1, xT, CIN);
  k_adjP<<<B * K, 256, 0, stream>>>(imp, adjPh);

  // layer 1
  k_gemm_sim<<<dim3(K / 64, K / 64, B), 128, 0, stream>>>(nT1, sim, CIN);
  k_thr_asum<<<(B * K) / 8, 256, 0, stream>>>(sim, adjPh, Asumh);
  k_gemm_h<<<dim3(K / 64, CMID / 64, B), 128, 0, stream>>>(W1t, xT, h1h, CMID, CIN);
  k_gemm_emb1<<<dim3(K / 64, CMID / 64, B), 128, 0, stream>>>(h1h, Asumh, emb1, CMID);

  // derive layer-2 inputs
  k_colnorm<<<(B * K) / 256, 256, 0, stream>>>(emb1, invn2, CMID);
  k_cast_T<<<dim3(K / 32, CMID / 32, B), dim3(32, 8), 0, stream>>>(emb1, invn2, kms, n2T, e1kT, CMID);
  k_bags<<<B * CMID, 256, 0, stream>>>(emb1, imp, topm, keptm, dropm, bagk, bagd);

  // layer 2
  k_gemm_sim<<<dim3(K / 64, K / 64, B), 128, 0, stream>>>(n2T, sim, CMID);
  k_thr_asum<<<(B * K) / 8, 256, 0, stream>>>(sim, adjPh, Asumh);
  k_gemm_h<<<dim3(K / 64, COUT / 64, B), 128, 0, stream>>>(W2t, e1kT, h2h, COUT, CMID);
  k_gemm_emb2<<<dim3(K / 64, COUT / 64, B), 128, 0, stream>>>(h2h, Asumh, bmask, imp, em2w, COUT);
  k_bagred<<<B * COUT, 256, 0, stream>>>(em2w, bag);

  // head
  k_pooled<<<B * CIN, 256, 0, stream>>>(x, imp, pooled);
  k_head<<<B, COUT, 0, stream>>>(pooled, bag, idw, idb, fcw, out);
}
